// GCNClassifier_44152263803370
// MI455X (gfx1250) — matrix operand order repaired and checked by reading, not compiled, not run on hardware
//
#include <hip/hip_runtime.h>
#include <hip/hip_bf16.h>
#include <math.h>

// ---------------------------------------------------------------------------
// GCN classifier for MI455X (gfx1250), wave32.
// WMMA bf16 GEMMs with weight fragments staged to LDS via CDNA5 async
// global->LDS copies (ASYNCcnt), f32-atomic edge aggregation.
// ---------------------------------------------------------------------------

typedef __attribute__((ext_vector_type(16))) __bf16 v16bf;
typedef __attribute__((ext_vector_type(8)))  float  v8f;

#define IN_DIM  384
#define HID     128

// ------------------------------ small utility kernels ----------------------

__global__ void k_fill(float* p, float v, size_t n) {
    size_t i = (size_t)blockIdx.x * blockDim.x + threadIdx.x;
    if (i < n) p[i] = v;
}

__global__ void k_deg_edges(const int* dst, float* deg, int E) {
    int e = blockIdx.x * blockDim.x + threadIdx.x;
    if (e < E)
        __hip_atomic_fetch_add(&deg[dst[e]], 1.0f,
                               __ATOMIC_RELAXED, __HIP_MEMORY_SCOPE_AGENT);
}

__global__ void k_dinv(float* deg, int N) {   // in place: deg -> rsqrt(deg)
    int i = blockIdx.x * blockDim.x + threadIdx.x;
    if (i < N) deg[i] = rsqrtf(deg[i]);       // deg >= 1 (self loop)
}

__global__ void k_bias_init(float* h, const float* b, int N) {
    size_t i = (size_t)blockIdx.x * blockDim.x + threadIdx.x;
    if (i < (size_t)N * HID) h[i] = b[i & (HID - 1)];
}

__global__ void k_relu(float* h, int N) {
    size_t i = (size_t)blockIdx.x * blockDim.x + threadIdx.x;
    if (i < (size_t)N * HID) h[i] = fmaxf(h[i], 0.0f);
}

// ------------------ W -> bf16 WMMA B-fragment repack -----------------------
// B frag (16-bit, 32x16 KxN): lane%16 = column, lane/16 picks K half
// (k = kt*32 + (lane/16)*16 + i). Each lane's 16 bf16 (32B) contiguous.
__global__ void k_wfrag(const float* __restrict__ W, __bf16* __restrict__ out,
                        int KT /* K/32 */) {
    int kt   = blockIdx.x >> 3;
    int nt   = blockIdx.x & 7;
    int lane = threadIdx.x;            // 0..31
    int n    = nt * 16 + (lane & 15);
    int kb   = kt * 32 + (lane >> 4) * 16;
    __bf16* o = out + ((size_t)(kt * 8 + nt) * 32 + lane) * 16;
    for (int i = 0; i < 16; ++i)
        o[i] = (__bf16)W[(size_t)(kb + i) * HID + n];
    (void)KT;
}

// ------------------------------- WMMA GEMM ---------------------------------
// out[N,128] = A[N,K] @ W (bf16 frags). One wave per 16-row strip, 8 waves
// per block. Weight fragments staged to LDS with async global->LDS copies,
// then the inner loop is A-stream (global) + B (LDS broadcast) + wmma.
template <int KT>
__global__ void __launch_bounds__(256)
k_gemm(const float* __restrict__ A, const __bf16* __restrict__ wfrag,
       float* __restrict__ out, int N, int K) {
    __shared__ v16bf sWf[KT * 8 * 32];          // KT*8KB

    const int tid  = threadIdx.x;
    const int lane = tid & 31;
    const int wave = tid >> 5;
    const int half = lane >> 4;
    const int m0   = (blockIdx.x * 8 + wave) * 16;

    // ---- async copy of weight fragments: global -> LDS (ASYNCcnt) ----
    {
        const int nchunks = KT * 512;           // 16B chunks
        const char* gbase = (const char*)wfrag;
        uint32_t    lbase = (uint32_t)(uintptr_t)(&sWf[0]);
        for (int i = tid; i < nchunks; i += 256) {
            uint64_t g = (uint64_t)(uintptr_t)(gbase + (size_t)i * 16);
            uint32_t l = lbase + (uint32_t)i * 16;
            asm volatile("global_load_async_to_lds_b128 %0, %1, off"
                         :: "v"(l), "v"(g) : "memory");
        }
        asm volatile("s_wait_asynccnt 0x0" ::: "memory");
    }
    __syncthreads();

    if (m0 < N) {
        int m = m0 + (lane & 15);
        if (m >= N) m = N - 1;                  // clamp loads; stores guarded
        const float* __restrict__ arow = A + (size_t)m * K;

        v8f acc[8];
        #pragma unroll
        for (int t = 0; t < 8; ++t) acc[t] = (v8f)0.0f;

        for (int kt = 0; kt < KT; ++kt) {
            // A frag: k = kt*32 + half*16 + i -> one contiguous 16-float run/lane
            const float* pa = arow + kt * 32 + half * 16;
            float4 r0 = *(const float4*)(pa + 0);
            float4 r1 = *(const float4*)(pa + 4);
            float4 r2 = *(const float4*)(pa + 8);
            float4 r3 = *(const float4*)(pa + 12);
            float fa[16] = { r0.x, r0.y, r0.z, r0.w, r1.x, r1.y, r1.z, r1.w,
                             r2.x, r2.y, r2.z, r2.w, r3.x, r3.y, r3.z, r3.w };
            v16bf a;
            #pragma unroll
            for (int i = 0; i < 16; ++i) a[i] = (__bf16)fa[i];

            const v16bf* bb = &sWf[kt * 8 * 32 + lane];
            #pragma unroll
            for (int nt = 0; nt < 8; ++nt) {
                v16bf b = bb[nt * 32];          // ds_load, broadcast to 8 waves
                acc[nt] = __builtin_amdgcn_wmma_f32_16x16x32_bf16(
                    false, a, false, b, (short)0, acc[nt], false, false);
            }
        }

        // D layout: VGPR r -> row m0 + half*8 + r, col = nt*16 + lane%16
        #pragma unroll
        for (int nt = 0; nt < 8; ++nt) {
            #pragma unroll
            for (int r = 0; r < 8; ++r) {
                int mr = m0 + half * 8 + r;
                if (mr < N)
                    out[(size_t)mr * HID + nt * 16 + (lane & 15)] = acc[nt][r];
            }
        }
    }
}

// --------------------- edge + self-loop scatter-add ------------------------
__global__ void k_agg(const float* __restrict__ t, float* __restrict__ h,
                      const int* __restrict__ src, const int* __restrict__ dst,
                      const float* __restrict__ dinv, int E, int N) {
    long long e = (long long)blockIdx.x * (blockDim.x >> 5) + (threadIdx.x >> 5);
    if (e >= (long long)E + N) return;
    int lane = threadIdx.x & 31;
    int s, d;
    if (e < E) { s = src[e]; d = dst[e]; }
    else       { s = d = (int)(e - E); }
    float w = dinv[s] * dinv[d];
    float4 v = *(const float4*)(t + (size_t)s * HID + lane * 4);
    float* hp = h + (size_t)d * HID + lane * 4;
    __hip_atomic_fetch_add(hp + 0, w * v.x, __ATOMIC_RELAXED, __HIP_MEMORY_SCOPE_AGENT);
    __hip_atomic_fetch_add(hp + 1, w * v.y, __ATOMIC_RELAXED, __HIP_MEMORY_SCOPE_AGENT);
    __hip_atomic_fetch_add(hp + 2, w * v.z, __ATOMIC_RELAXED, __HIP_MEMORY_SCOPE_AGENT);
    __hip_atomic_fetch_add(hp + 3, w * v.w, __ATOMIC_RELAXED, __HIP_MEMORY_SCOPE_AGENT);
}

// ------------------------------ mean pool ----------------------------------
__global__ void k_pool(const float* __restrict__ h, const int* __restrict__ batch,
                       float* __restrict__ pooled, float* __restrict__ counts, int N) {
    long long node = (long long)blockIdx.x * (blockDim.x >> 5) + (threadIdx.x >> 5);
    if (node >= N) return;
    int lane = threadIdx.x & 31;
    int g = batch[node];
    float4 v = *(const float4*)(h + (size_t)node * HID + lane * 4);
    float* pp = pooled + (size_t)g * HID + lane * 4;
    __hip_atomic_fetch_add(pp + 0, v.x, __ATOMIC_RELAXED, __HIP_MEMORY_SCOPE_AGENT);
    __hip_atomic_fetch_add(pp + 1, v.y, __ATOMIC_RELAXED, __HIP_MEMORY_SCOPE_AGENT);
    __hip_atomic_fetch_add(pp + 2, v.z, __ATOMIC_RELAXED, __HIP_MEMORY_SCOPE_AGENT);
    __hip_atomic_fetch_add(pp + 3, v.w, __ATOMIC_RELAXED, __HIP_MEMORY_SCOPE_AGENT);
    if (lane == 0)
        __hip_atomic_fetch_add(&counts[g], 1.0f,
                               __ATOMIC_RELAXED, __HIP_MEMORY_SCOPE_AGENT);
}

// ------------------------ linear head + log-softmax ------------------------
__global__ void k_head(const float* __restrict__ pooled, const float* __restrict__ counts,
                       const float* __restrict__ fc_w, const float* __restrict__ fc_b,
                       float* __restrict__ out, int G) {
    int g = blockIdx.x * blockDim.x + threadIdx.x;
    if (g >= G) return;
    float inv = 1.0f / fmaxf(counts[g], 1.0f);
    float z0 = fc_b[0], z1 = fc_b[1];
    const float* p = pooled + (size_t)g * HID;
    for (int c = 0; c < HID; ++c) {
        float v = p[c] * inv;
        z0 += v * fc_w[c * 2 + 0];
        z1 += v * fc_w[c * 2 + 1];
    }
    float mx  = fmaxf(z0, z1);
    float lse = mx + logf(expf(z0 - mx) + expf(z1 - mx));
    out[g * 2 + 0] = z0 - lse;
    out[g * 2 + 1] = z1 - lse;
}

// ---------------------------------------------------------------------------

static inline size_t alignUp(size_t x) { return (x + 255) & ~(size_t)255; }

extern "C" void kernel_launch(void* const* d_in, const int* in_sizes, int n_in,
                              void* d_out, int out_size, void* d_ws, size_t ws_size,
                              hipStream_t stream) {
    const float* x    = (const float*)d_in[0];
    const int*   ei   = (const int*)d_in[1];
    const int*   bat  = (const int*)d_in[2];
    const float* W1   = (const float*)d_in[3];
    const float* b1   = (const float*)d_in[4];
    const float* W2   = (const float*)d_in[5];
    const float* b2   = (const float*)d_in[6];
    const float* fcw  = (const float*)d_in[7];
    const float* fcb  = (const float*)d_in[8];
    float*       out  = (float*)d_out;

    const int N = in_sizes[0] / IN_DIM;     // 100000
    const int E = in_sizes[1] / 2;          // 1600000
    const int G = out_size / 2;             // 512
    const int* src = ei;
    const int* dst = ei + E;

    // -------- workspace carve-up (bytes) --------
    char*  w  = (char*)d_ws;
    size_t o  = 0;
    float* dinv   = (float*)(w + o); o = alignUp(o + (size_t)N * 4);
    float* buf0   = (float*)(w + o); o = alignUp(o + (size_t)N * HID * 4);
    float* buf1   = (float*)(w + o); o = alignUp(o + (size_t)N * HID * 4);
    float* pooled = (float*)(w + o); o = alignUp(o + (size_t)G * HID * 4);
    float* counts = (float*)(w + o); o = alignUp(o + (size_t)G * 4);
    __bf16* wf1   = (__bf16*)(w + o); o = alignUp(o + (size_t)12 * 8 * 32 * 16 * 2);
    __bf16* wf2   = (__bf16*)(w + o); o = alignUp(o + (size_t)4  * 8 * 32 * 16 * 2);
    (void)ws_size; (void)n_in;

    const size_t NH = (size_t)N * HID;
    const int  TB = 256;
    const int  gNH   = (int)((NH + TB - 1) / TB);
    const int  gN    = (N + TB - 1) / TB;
    const int  gE    = (E + TB - 1) / TB;
    const long long ET = (long long)E + N;
    const int  gAgg  = (int)((ET + 7) / 8);          // 8 edges (waves) / block
    const int  gPool = (N + 7) / 8;                  // 8 nodes / block
    const int  gGemm = (N + 127) / 128;              // 128 rows / block

    // -------- degrees + normalization --------
    k_fill<<<gN, TB, 0, stream>>>(dinv, 1.0f, (size_t)N);        // self loop
    k_deg_edges<<<gE, TB, 0, stream>>>(dst, dinv, E);
    k_dinv<<<gN, TB, 0, stream>>>(dinv, N);

    // -------- weight repack to bf16 B-fragments --------
    k_wfrag<<<12 * 8, 32, 0, stream>>>(W1, wf1, 12);
    k_wfrag<<< 4 * 8, 32, 0, stream>>>(W2, wf2, 4);

    // -------- layer 1: t = x@W1 ; h = relu(agg(t) + b1) --------
    k_gemm<12><<<gGemm, TB, 0, stream>>>(x, wf1, buf0, N, IN_DIM);
    k_bias_init<<<gNH, TB, 0, stream>>>(buf1, b1, N);
    k_agg<<<gAgg, TB, 0, stream>>>(buf0, buf1, src, dst, dinv, E, N);
    k_relu<<<gNH, TB, 0, stream>>>(buf1, N);

    // -------- layer 2: t = h@W2 ; h = relu(agg(t) + b2) --------
    k_gemm<4><<<gGemm, TB, 0, stream>>>(buf1, wf2, buf0, N, HID);
    k_bias_init<<<gNH, TB, 0, stream>>>(buf1, b2, N);
    k_agg<<<gAgg, TB, 0, stream>>>(buf0, buf1, src, dst, dinv, E, N);
    k_relu<<<gNH, TB, 0, stream>>>(buf1, N);

    // -------- global mean pool + head --------
    k_fill<<<(int)(((size_t)G * HID + TB - 1) / TB), TB, 0, stream>>>(pooled, 0.0f, (size_t)G * HID);
    k_fill<<<(G + TB - 1) / TB, TB, 0, stream>>>(counts, 0.0f, (size_t)G);
    k_pool<<<gPool, TB, 0, stream>>>(buf1, bat, pooled, counts, N);
    k_head<<<(G + TB - 1) / TB, TB, 0, stream>>>(pooled, counts, fcw, fcb, out, G);
}
